// MultiHeadAttention_3951369912390
// MI455X (gfx1250) — matrix operand order repaired and checked by reading, not compiled, not run on hardware
//
#include <hip/hip_runtime.h>
#include <hip/hip_bf16.h>

#define BB 4
#define LSEQ 2048
#define DM 1024
#define NH 16
#define DK 64
#define NEG_FILL -1e9f
#define LN_EPS 1e-6f

typedef __attribute__((ext_vector_type(16))) __bf16 v16bf;
typedef __attribute__((ext_vector_type(8)))  float  v8f;
typedef __attribute__((ext_vector_type(4)))  int    v4i;

union FragBF { v16bf v; unsigned int d[8]; };

#if defined(__HIP_DEVICE_COMPILE__) && __has_builtin(__builtin_amdgcn_global_load_async_to_lds_b128) && __has_builtin(__builtin_amdgcn_s_wait_asynccnt)
#define HAVE_ASYNC_LDS 1
#else
#define HAVE_ASYNC_LDS 0
#endif

#if HAVE_ASYNC_LDS
typedef __attribute__((address_space(1))) v4i* gv4i_p;  // global <4 x i32>*
typedef __attribute__((address_space(3))) v4i* lv4i_p;  // LDS <4 x i32>*
#endif

__device__ __forceinline__ void async_b128(unsigned short* lds_dst, const unsigned short* gsrc) {
#if HAVE_ASYNC_LDS
  __builtin_amdgcn_global_load_async_to_lds_b128(
      (gv4i_p)(void*)gsrc,
      (lv4i_p)(void*)lds_dst,
      0, 0);
#else
  *(uint4*)lds_dst = *(const uint4*)gsrc;
#endif
}

__device__ __forceinline__ void wait_async() {
#if HAVE_ASYNC_LDS
  __builtin_amdgcn_s_wait_asynccnt(0);
#endif
}

__device__ __forceinline__ unsigned short f32_to_bf16(float f) {
  unsigned int u = __float_as_uint(f);
  unsigned int r = 0x7FFFu + ((u >> 16) & 1u);
  return (unsigned short)((u + r) >> 16);
}

// A-matrix 16x32 bf16 fragment from LDS tile stored row-major [m][k] (stride in ushorts).
// ISA layout: lanes 0-15 M=lane, lanes 16-31 M=lane-16; VGPR0-3: K=2v+p+8*half,
// VGPR4-7: K=16+2(v-4)+p+8*half  -> contiguous dword LDS reads per lane.
__device__ __forceinline__ v16bf load_a_frag(const unsigned short* lds, int m_base, int stride) {
  const int lane = (int)(threadIdx.x & 31u);
  const int half = lane >> 4;
  const unsigned short* row = lds + (m_base + (lane & 15)) * stride;
  FragBF f;
#pragma unroll
  for (int j = 0; j < 4; ++j) {
    f.d[j]     = *(const unsigned int*)(row + 2 * j + 8 * half);
    f.d[j + 4] = *(const unsigned int*)(row + 16 + 2 * j + 8 * half);
  }
  return f.v;
}

// B-matrix 32x16 bf16 fragment from LDS tile stored transposed [n][k] (stride in ushorts).
// lanes 0-15: N=lane, lanes 16-31: N=lane-16; dword j holds K=2(j&3)+8*half+16*(j>>2) (+0,+1), matching A.
__device__ __forceinline__ v16bf load_b_frag(const unsigned short* lds, int n_base, int stride) {
  const int lane = (int)(threadIdx.x & 31u);
  const int half = lane >> 4;
  const unsigned short* row = lds + (n_base + (lane & 15)) * stride;
  FragBF f;
#pragma unroll
  for (int j = 0; j < 8; ++j)
    f.d[j] = *(const unsigned int*)(row + 2 * (j & 3) + 8 * half + 16 * (j >> 2));
  return f.v;
}

__device__ __forceinline__ v8f wmma_bf16(v16bf a, v16bf b, v8f c) {
  return __builtin_amdgcn_wmma_f32_16x16x32_bf16(false, a, false, b, (short)0, c, false, false);
}

// ---------------- LayerNorm(q) -> bf16 ----------------
__global__ void ln_kernel(const float* __restrict__ q, const float* __restrict__ gamma,
                          const float* __restrict__ beta, unsigned short* __restrict__ qn) {
  __shared__ float red[256];
  const int row = blockIdx.x;
  const int tid = threadIdx.x;
  const float* x = q + (size_t)row * DM;
  float x4[4];
  float s = 0.f;
#pragma unroll
  for (int j = 0; j < 4; ++j) { x4[j] = x[tid + 256 * j]; s += x4[j]; }
  red[tid] = s; __syncthreads();
  for (int o = 128; o > 0; o >>= 1) { if (tid < o) red[tid] += red[tid + o]; __syncthreads(); }
  const float mu = red[0] * (1.0f / DM);
  __syncthreads();
  float vv = 0.f;
#pragma unroll
  for (int j = 0; j < 4; ++j) { float d = x4[j] - mu; vv += d * d; }
  red[tid] = vv; __syncthreads();
  for (int o = 128; o > 0; o >>= 1) { if (tid < o) red[tid] += red[tid + o]; __syncthreads(); }
  const float rstd = rsqrtf(red[0] * (1.0f / DM) + LN_EPS);
#pragma unroll
  for (int j = 0; j < 4; ++j) {
    int c = tid + 256 * j;
    qn[(size_t)row * DM + c] = f32_to_bf16((x4[j] - mu) * rstd * gamma[c] + beta[c]);
  }
}

// ---------------- f32 -> bf16 convert (activations) ----------------
__global__ void cvt_kernel(const float* __restrict__ in, unsigned short* __restrict__ out, int n) {
  int i = blockIdx.x * 256 + threadIdx.x;
  if (i < n) out[i] = f32_to_bf16(in[i]);
}

// ---------------- f32 [K,N] -> bf16 transposed [N,K] (weights) ----------------
__global__ void cvt_t_kernel(const float* __restrict__ in, unsigned short* __restrict__ out) {
  __shared__ unsigned short tile[32][33];
  const int kx = blockIdx.x * 32;
  const int nx = blockIdx.y * 32;
  const int tx = threadIdx.x & 31, ty = threadIdx.x >> 5;  // 32 x 8
#pragma unroll
  for (int j = 0; j < 4; ++j)
    tile[ty + 8 * j][tx] = f32_to_bf16(in[(size_t)(kx + ty + 8 * j) * DM + nx + tx]);
  __syncthreads();
#pragma unroll
  for (int j = 0; j < 4; ++j)
    out[(size_t)(nx + ty + 8 * j) * DM + kx + tx] = tile[tx][ty + 8 * j];
}

// ---------------- WMMA GEMM: [M=8192,1024] x Wt[N=1024,K=1024] ----------------
// MODE 0: bf16 head-major [B,H,L,DK].  MODE 1: f32 [M,DM] + residual.
// MODE 2: bf16 transposed head-major [B,H,DK,L] (for V).
template<int MODE>
__global__ void wmma_gemm_kernel(const unsigned short* __restrict__ X,
                                 const unsigned short* __restrict__ Wt,
                                 unsigned short* __restrict__ Ybf,
                                 float* __restrict__ Yf,
                                 const float* __restrict__ residual) {
  __shared__ __align__(16) unsigned short lsA[64 * 64];
  __shared__ __align__(16) unsigned short lsB[64 * 64];
  const int tid = threadIdx.x;
  const int lane = tid & 31, half = lane >> 4, wave = tid >> 5;
  const int m_sub = (wave >> 2) * 16;
  const int n_sub = (wave & 3) * 16;
  const int n0 = blockIdx.x * 64;
  const int m0 = blockIdx.y * 64;

  const int lin2 = tid * 16;
  const int ar = lin2 >> 6, ac = lin2 & 63;   // 64 rows x 64 k-chunk, 16 ushorts/thread

  v8f acc0 = {}, acc1 = {};

  for (int kk = 0; kk < DM; kk += 64) {
    async_b128(lsA + lin2,     X  + (size_t)(m0 + ar) * DM + kk + ac);
    async_b128(lsA + lin2 + 8, X  + (size_t)(m0 + ar) * DM + kk + ac + 8);
    async_b128(lsB + lin2,     Wt + (size_t)(n0 + ar) * DM + kk + ac);
    async_b128(lsB + lin2 + 8, Wt + (size_t)(n0 + ar) * DM + kk + ac + 8);
    wait_async();
    __syncthreads();
#pragma unroll
    for (int ks = 0; ks < 64; ks += 32) {
      v16bf bf = load_b_frag(lsB + ks, n_sub, 64);
      v16bf a0 = load_a_frag(lsA + ks, m_sub, 64);
      v16bf a1 = load_a_frag(lsA + ks, m_sub + 32, 64);
      acc0 = wmma_bf16(a0, bf, acc0);
      acc1 = wmma_bf16(a1, bf, acc1);
    }
    __syncthreads();
  }

#pragma unroll
  for (int r = 0; r < 8; ++r) {
    const int mrow0 = m0 + m_sub + r + 8 * half;
    const int mrow1 = mrow0 + 32;
    const int ncol = n0 + n_sub + (lane & 15);
    if (MODE == 0) {
      const int h = ncol >> 6, d = ncol & 63;
      { int b = mrow0 / LSEQ, l = mrow0 % LSEQ;
        Ybf[(((size_t)b * NH + h) * LSEQ + l) * DK + d] = f32_to_bf16(acc0[r]); }
      { int b = mrow1 / LSEQ, l = mrow1 % LSEQ;
        Ybf[(((size_t)b * NH + h) * LSEQ + l) * DK + d] = f32_to_bf16(acc1[r]); }
    } else if (MODE == 2) {
      const int h = ncol >> 6, d = ncol & 63;
      { int b = mrow0 / LSEQ, l = mrow0 % LSEQ;
        Ybf[(((size_t)b * NH + h) * DK + d) * LSEQ + l] = f32_to_bf16(acc0[r]); }
      { int b = mrow1 / LSEQ, l = mrow1 % LSEQ;
        Ybf[(((size_t)b * NH + h) * DK + d) * LSEQ + l] = f32_to_bf16(acc1[r]); }
    } else {
      Yf[(size_t)mrow0 * DM + ncol] = residual[(size_t)mrow0 * DM + ncol] + acc0[r];
      Yf[(size_t)mrow1 * DM + ncol] = residual[(size_t)mrow1 * DM + ncol] + acc1[r];
    }
  }
}

// ---------------- scores = QK^T / 8 with causal mask -> d_out ----------------
__global__ void scores_kernel(const unsigned short* __restrict__ qh,
                              const unsigned short* __restrict__ kh,
                              float* __restrict__ scores) {
  const int k0 = blockIdx.x * 64;
  const int q0 = blockIdx.y * 64;
  const int bh = blockIdx.z;
  const int tid = threadIdx.x;
  float* stile = scores + (size_t)bh * LSEQ * LSEQ;

  if (k0 > q0 + 63) {  // fully above diagonal: fill only
    for (int i = tid; i < 64 * 64; i += 256) {
      int m = i >> 6, n = i & 63;
      stile[(size_t)(q0 + m) * LSEQ + k0 + n] = NEG_FILL;
    }
    return;
  }

  __shared__ __align__(16) unsigned short lsQ[64 * 64];
  __shared__ __align__(16) unsigned short lsK[64 * 64];
  {
    const int lin = tid * 16;
    const int m = lin >> 6, c = lin & 63;
    async_b128(lsQ + lin,     qh + ((size_t)bh * LSEQ + q0 + m) * DK + c);
    async_b128(lsQ + lin + 8, qh + ((size_t)bh * LSEQ + q0 + m) * DK + c + 8);
    async_b128(lsK + lin,     kh + ((size_t)bh * LSEQ + k0 + m) * DK + c);
    async_b128(lsK + lin + 8, kh + ((size_t)bh * LSEQ + k0 + m) * DK + c + 8);
  }
  wait_async();
  __syncthreads();

  const int lane = tid & 31, half = lane >> 4, wave = tid >> 5;
  const int m_sub = (wave >> 2) * 16, n_sub = (wave & 3) * 16;
  v8f acc0 = {}, acc1 = {};
#pragma unroll
  for (int ks = 0; ks < 64; ks += 32) {
    v16bf bf = load_b_frag(lsK + ks, n_sub, 64);   // kh row-major == K^T in [n][k] form
    v16bf a0 = load_a_frag(lsQ + ks, m_sub, 64);
    v16bf a1 = load_a_frag(lsQ + ks, m_sub + 32, 64);
    acc0 = wmma_bf16(a0, bf, acc0);
    acc1 = wmma_bf16(a1, bf, acc1);
  }
#pragma unroll
  for (int r = 0; r < 8; ++r) {
    const int qr0 = q0 + m_sub + r + 8 * half;
    const int qr1 = qr0 + 32;
    const int kc = k0 + n_sub + (lane & 15);
    stile[(size_t)qr0 * LSEQ + kc] = (kc <= qr0) ? acc0[r] * 0.125f : NEG_FILL;
    stile[(size_t)qr1 * LSEQ + kc] = (kc <= qr1) ? acc1[r] * 0.125f : NEG_FILL;
  }
}

// ---------------- per-row softmax stats ----------------
__global__ void softmax_stats_kernel(const float* __restrict__ scores,
                                     float* __restrict__ rmax, float* __restrict__ rinv) {
  __shared__ float red[256];
  const int row = blockIdx.x;
  const int tid = threadIdx.x;
  const float* p = scores + (size_t)row * LSEQ;
  float m = -3.0e38f;
#pragma unroll
  for (int j = 0; j < LSEQ / 256; ++j) m = fmaxf(m, p[tid + 256 * j]);
  red[tid] = m; __syncthreads();
  for (int o = 128; o > 0; o >>= 1) { if (tid < o) red[tid] = fmaxf(red[tid], red[tid + o]); __syncthreads(); }
  m = red[0]; __syncthreads();
  float s = 0.f;
#pragma unroll
  for (int j = 0; j < LSEQ / 256; ++j) s += __expf(p[tid + 256 * j] - m);
  red[tid] = s; __syncthreads();
  for (int o = 128; o > 0; o >>= 1) { if (tid < o) red[tid] += red[tid + o]; __syncthreads(); }
  if (tid == 0) { rmax[row] = m; rinv[row] = 1.0f / red[0]; }
}

// ---------------- x = softmax(scores) @ V (flash-style, causal-trimmed) ----------------
// vt is [B,H,DK,L] so V staging is contiguous b128 loads.
__global__ void pv_kernel(const float* __restrict__ scores,
                          const float* __restrict__ rmax, const float* __restrict__ rinv,
                          const unsigned short* __restrict__ vt,
                          unsigned short* __restrict__ xout) {
  __shared__ __align__(16) unsigned short lsP[64 * 32];
  __shared__ __align__(16) unsigned short lsV[64 * 32];
  const int q0 = blockIdx.x * 64;
  const int bh = blockIdx.y;
  const int b = bh >> 4, h = bh & 15;
  const int tid = threadIdx.x;
  const int lane = tid & 31, half = lane >> 4, wave = tid >> 5;
  const int m_sub = (wave >> 2) * 16, n_sub = (wave & 3) * 16;

  const int lin = tid * 8;
  const int pm = lin >> 5, pk = lin & 31;      // P: row 0..63, k-chunk {0,8,16,24}
  const int prow = q0 + pm;
  const float row_max = rmax[(size_t)bh * LSEQ + prow];
  const float row_inv = rinv[(size_t)bh * LSEQ + prow];
  const float* srow = scores + ((size_t)bh * LSEQ + prow) * LSEQ;
  const unsigned short* vrow = vt + ((size_t)bh * DK + pm) * LSEQ;  // pm doubles as dv

  v8f acc0 = {}, acc1 = {};
  const int kend = q0 + 64;            // causal: columns past the tile diagonal are exp(-1e9)=0
  for (int kk = 0; kk < kend; kk += 32) {
    async_b128(lsV + lin, vrow + kk + pk);       // lsV[dv][kpos]
#pragma unroll
    for (int j = 0; j < 8; ++j)                  // build P chunk in bf16
      lsP[pm * 32 + pk + j] = f32_to_bf16(__expf(srow[kk + pk + j] - row_max) * row_inv);
    wait_async();
    __syncthreads();
    v16bf bf = load_b_frag(lsV, n_sub, 32);
    v16bf a0 = load_a_frag(lsP, m_sub, 32);
    v16bf a1 = load_a_frag(lsP, m_sub + 32, 32);
    acc0 = wmma_bf16(a0, bf, acc0);
    acc1 = wmma_bf16(a1, bf, acc1);
    __syncthreads();
  }
#pragma unroll
  for (int r = 0; r < 8; ++r) {
    const int q0r = q0 + m_sub + r + 8 * half;
    const int q1r = q0r + 32;
    const int dv = n_sub + (lane & 15);
    xout[((size_t)(b * LSEQ + q0r)) * DM + h * DK + dv] = f32_to_bf16(acc0[r]);
    xout[((size_t)(b * LSEQ + q1r)) * DM + h * DK + dv] = f32_to_bf16(acc1[r]);
  }
}

extern "C" void kernel_launch(void* const* d_in, const int* in_sizes, int n_in,
                              void* d_out, int out_size, void* d_ws, size_t ws_size,
                              hipStream_t stream) {
  const float* q  = (const float*)d_in[0];
  const float* k  = (const float*)d_in[1];
  const float* v  = (const float*)d_in[2];
  const float* Wq = (const float*)d_in[3];
  const float* Wk = (const float*)d_in[4];
  const float* Wv = (const float*)d_in[5];
  const float* Wo = (const float*)d_in[6];
  const float* gamma = (const float*)d_in[7];
  const float* beta  = (const float*)d_in[8];

  float* out_f  = (float*)d_out;                         // [B,L,DM]
  float* scores = out_f + (size_t)BB * LSEQ * DM;        // [B,H,L,L]

  char* ws = (char*)d_ws;
  size_t off = 0;
  auto walloc = [&](size_t bytes) -> char* {
    char* p = ws + off; off += (bytes + 255) & ~(size_t)255; return p;
  };
  const size_t actN = (size_t)BB * LSEQ * DM;            // 8.39M elements
  const size_t wN   = (size_t)DM * DM;                   // 1.05M elements
  unsigned short* qn   = (unsigned short*)walloc(actN * 2);
  unsigned short* kb   = (unsigned short*)walloc(actN * 2);
  unsigned short* vb   = (unsigned short*)walloc(actN * 2);
  unsigned short* Wqt  = (unsigned short*)walloc(wN * 2);
  unsigned short* Wkt  = (unsigned short*)walloc(wN * 2);
  unsigned short* Wvt  = (unsigned short*)walloc(wN * 2);
  unsigned short* Wot  = (unsigned short*)walloc(wN * 2);
  unsigned short* qh   = (unsigned short*)walloc(actN * 2);
  unsigned short* kh   = (unsigned short*)walloc(actN * 2);
  unsigned short* vt   = (unsigned short*)walloc(actN * 2);
  unsigned short* xo   = (unsigned short*)walloc(actN * 2);
  float* rmax = (float*)walloc((size_t)BB * NH * LSEQ * 4);
  float* rinv = (float*)walloc((size_t)BB * NH * LSEQ * 4);

  // 1. LayerNorm(q) -> bf16
  ln_kernel<<<BB * LSEQ, 256, 0, stream>>>(q, gamma, beta, qn);
  // 2. bf16 converts (activations) + transposed bf16 weights [N,K]
  cvt_kernel<<<(int)(actN / 256), 256, 0, stream>>>(k, kb, (int)actN);
  cvt_kernel<<<(int)(actN / 256), 256, 0, stream>>>(v, vb, (int)actN);
  dim3 tgrid(DM / 32, DM / 32);
  cvt_t_kernel<<<tgrid, 256, 0, stream>>>(Wq, Wqt);
  cvt_t_kernel<<<tgrid, 256, 0, stream>>>(Wk, Wkt);
  cvt_t_kernel<<<tgrid, 256, 0, stream>>>(Wv, Wvt);
  cvt_t_kernel<<<tgrid, 256, 0, stream>>>(Wo, Wot);

  // 3. Q/K/V projections
  dim3 ggrid(DM / 64, (BB * LSEQ) / 64);
  wmma_gemm_kernel<0><<<ggrid, 256, 0, stream>>>(qn, Wqt, qh, nullptr, nullptr);
  wmma_gemm_kernel<0><<<ggrid, 256, 0, stream>>>(kb, Wkt, kh, nullptr, nullptr);
  wmma_gemm_kernel<2><<<ggrid, 256, 0, stream>>>(vb, Wvt, vt, nullptr, nullptr);

  // 4. masked scores -> d_out
  scores_kernel<<<dim3(LSEQ / 64, LSEQ / 64, BB * NH), 256, 0, stream>>>(qh, kh, scores);

  // 5. softmax row stats
  softmax_stats_kernel<<<BB * NH * LSEQ, 256, 0, stream>>>(scores, rmax, rinv);

  // 6. attention @ V (causal-trimmed)
  pv_kernel<<<dim3(LSEQ / 64, BB * NH), 256, 0, stream>>>(scores, rmax, rinv, vt, xo);

  // 7. output projection + residual -> d_out
  wmma_gemm_kernel<1><<<ggrid, 256, 0, stream>>>(xo, Wot, nullptr, out_f, q);
}
